// HyperbolicDomainAdaptLayer_60404420051499
// MI455X (gfx1250) — matrix operand order repaired and checked by reading, not compiled, not run on hardware
//
#include <hip/hip_runtime.h>
#include <hip/hip_bf16.h>

#define NROI 116
#define PAD  128
#define NBATCH 512
#define MINN 1e-7f
#define MAXN 0.996f

typedef __attribute__((ext_vector_type(16))) __bf16 v16bf;
typedef __attribute__((ext_vector_type(8)))  float  v8f;

static __device__ __forceinline__ float artanh_(float x) {
    float c = fminf(fmaxf(x, -1.0f + 1e-5f), 1.0f - 1e-5f);
    return 0.5f * (log1pf(c) - log1pf(-c));
}
static __device__ __forceinline__ unsigned pack_bf16(float a, float b) {
    union { __bf16 h[2]; unsigned u; } p;
    p.h[0] = (__bf16)a; p.h[1] = (__bf16)b;
    return p.u;
}
static __device__ __forceinline__ float bf2f(unsigned u16) {
    union { unsigned u; float f; } c; c.u = u16 << 16; return c.f;
}
static __device__ __forceinline__ float sq8_from_u4(uint4 d) {
    unsigned w[4] = {d.x, d.y, d.z, d.w};
    float s = 0.f;
#pragma unroll
    for (int i = 0; i < 4; ++i) {
        float lo = bf2f(w[i] & 0xffffu), hi = bf2f(w[i] >> 16);
        s += lo * lo + hi * hi;
    }
    return s;
}

// ---- WMMA fragment builders (CDNA5 wave32 16x16x32 bf16 layouts) ----
// A tile staged as bf16[16][128]; frag for K-chunk k0.
static __device__ __forceinline__ v16bf make_a_frag128(const __bf16* Ald, int lane, int k0) {
    int m = lane & 15, h = lane >> 4;
    const __bf16* p = Ald + m * PAD + k0 + h * 8;
    v16bf a;
#pragma unroll
    for (int j = 0; j < 8; ++j) a[j] = p[j];
#pragma unroll
    for (int j = 0; j < 8; ++j) a[8 + j] = p[16 + j];
    return a;
}
// B (32x16): lane l -> n=l&15, h=l>>4; elem j<8 -> K = k0 + h*8 + j, elem 8+j -> K = k0 + 16 + h*8 + j
static __device__ __forceinline__ v16bf make_b_frag(const __bf16* Wld, int lane, int wave, int k0) {
    int n = (lane & 15) + wave * 16, h = lane >> 4;
    const __bf16* p = Wld + n * PAD + k0 + h * 8;
    v16bf b;
#pragma unroll
    for (int j = 0; j < 8; ++j) b[j] = p[j];
#pragma unroll
    for (int j = 0; j < 8; ++j) b[8 + j] = p[16 + j];
    return b;
}

// 16x128 row-tile: A is a contiguous 4KB bf16 span in global -> one b128 copy per thread,
// then 4 chained WMMAs; only two barriers per tile.
static __device__ __forceinline__ v8f gemm_tile_bf16(const __bf16* __restrict__ Atile,
                                                     const __bf16* Wld, __bf16* AldT,
                                                     float* xxOut, int tid, int wave, int lane) {
    __builtin_prefetch(Atile + 16 * PAD, 0, 1);         // next row-tile (contiguous)
    uint4 d = ((const uint4*)Atile)[tid];
    ((uint4*)AldT)[tid] = d;                            // row = tid>>4, 8 elems per thread
    if (xxOut) *xxOut = sq8_from_u4(d);                 // ||x row||^2 partial (one row/thread)
    __syncthreads();
    v8f acc = {0.f, 0.f, 0.f, 0.f, 0.f, 0.f, 0.f, 0.f};
#pragma unroll
    for (int ks = 0; ks < 4; ++ks) {
        v16bf af = make_a_frag128(AldT, lane, ks * 32);
        v16bf bf = make_b_frag(Wld, lane, wave, ks * 32);
        acc = __builtin_amdgcn_wmma_f32_16x16x32_bf16(false, af, false, bf, (short)0, acc, false, false);
    }
    __syncthreads();
    return acc;
}

// ---- LDS store-tree row reduction (no bpermute chains) ----
// redT: float[128*16] (col-major cells); red2: float[16*16]
static __device__ __forceinline__ void red_store(float* redT, const float* v, int c, int rbase) {
    float4 a = {v[0], v[1], v[2], v[3]}, b = {v[4], v[5], v[6], v[7]};
    *(float4*)(redT + c * 16 + rbase) = a;
    *(float4*)(redT + c * 16 + rbase + 4) = b;
}
static __device__ __forceinline__ void red_fold(const float* redT, float* red2, int tid) {
    int row = tid & 15, seg = tid >> 4;
    float s = 0.f;
#pragma unroll
    for (int j = 0; j < 8; ++j) s += redT[(seg * 8 + j) * 16 + row];
    red2[row * 16 + seg] = s;
}
static __device__ __forceinline__ float red_final(const float* red2, int row) {
    float s = 0.f;
#pragma unroll
    for (int j = 0; j < 16; ++j) s += red2[row * 16 + j];
    return s;
}

// ---- P0: pad x (B,116,116) f32 -> (B,128,128) bf16, zero fill ----
__global__ __launch_bounds__(256) void k_pad(const float* __restrict__ X, unsigned* __restrict__ XP) {
    size_t p = (size_t)blockIdx.x * 256 + threadIdx.x;   // pair index, total B*128*64
    int k2 = (int)(p & 63) * 2;
    int r  = (int)((p >> 6) & 127);
    int b  = (int)(p >> 13);
    float2 v = {0.f, 0.f};
    if (r < NROI && k2 < NROI) v = *(const float2*)(X + ((size_t)b * NROI + r) * NROI + k2);
    XP[p] = pack_bf16(v.x, v.y);
}

// ---- K0: hb = proj(expmap0(b)) for bq,bk,bv,bg (padded to 128) + |hb|^2; same for bhl ----
__global__ void k_prep(const float* bq, const float* bk, const float* bv, const float* bg,
                       const float* bhl, float* hbs, float* hbn2, float* hbl, float* hbln2) {
    int tid = threadIdx.x;
    if (tid < 4) {
        const float* src = (tid == 0) ? bq : (tid == 1) ? bk : (tid == 2) ? bv : bg;
        float s = 0.f;
        for (int i = 0; i < NROI; ++i) s += src[i] * src[i];
        float n = fmaxf(sqrtf(s), MINN);
        float c = fminf(tanhf(n), MAXN) / n;
        for (int i = 0; i < PAD; ++i) hbs[tid * PAD + i] = (i < NROI) ? c * src[i] : 0.f;
        float nn = fminf(tanhf(n), MAXN);
        hbn2[tid] = nn * nn;
    } else if (tid == 4) {
        float s = bhl[0] * bhl[0] + bhl[1] * bhl[1];
        float n = fmaxf(sqrtf(s), MINN);
        float c = fminf(tanhf(n), MAXN) / n;
        hbl[0] = c * bhl[0];
        hbl[1] = c * bhl[1];
        float nn = fminf(tanhf(n), MAXN);
        hbln2[0] = nn * nn;
    }
}

// ---- K1: hyp_linear rows (X padded bf16); bf16 or f32 output; optional trailing logmap0 ----
__global__ __launch_bounds__(256) void k_hyp_linear(const __bf16* __restrict__ X,
                                                    const float* __restrict__ W,
                                                    const float* __restrict__ hb,
                                                    const float* __restrict__ hbn2p,
                                                    void* __restrict__ outPtr,
                                                    int out_bf16, int do_logmap) {
    __shared__ __bf16 Wld[PAD * PAD];
    __shared__ __bf16 AldT[16 * PAD];
    __shared__ float redA[128 * 16], redB[128 * 16];
    __shared__ float red2A[256], red2B[256];
    __shared__ float rXX[16], sA[16], sB[16], sF[16];
    int tid = threadIdx.x, wave = tid >> 5, lane = tid & 31, blk = blockIdx.x;
    unsigned* WldU = (unsigned*)Wld;
    for (int i = tid; i < PAD * 64; i += 256) {          // W tight 116x116 f32, pair loads
        int n = i >> 6, k2 = (i & 63) * 2;
        float2 v = {0.f, 0.f};
        if (n < NROI && k2 < NROI) v = *(const float2*)(W + n * NROI + k2);
        WldU[i] = pack_bf16(v.x, v.y);
    }
    int cn = (lane & 15) + wave * 16;
    int rbase = (lane >> 4) * 8;
    int c = wave * 16 + (lane & 15);
    float hbv = hb[cn];
    float hbn2 = hbn2p[0];
    for (int rt = 0; rt < 8; ++rt) {
        const __bf16* Atile = X + ((size_t)blk * PAD + rt * 16) * PAD;
        float xx;
        v8f acc = gemm_tile_bf16(Atile, Wld, AldT, &xx, tid, wave, lane);
        xx += __shfl_xor(xx, 1); xx += __shfl_xor(xx, 2);
        xx += __shfl_xor(xx, 4); xx += __shfl_xor(xx, 8);
        if ((lane & 15) == 0) rXX[tid >> 4] = xx;
        float cv[8], t0[8], t1[8];
#pragma unroll
        for (int r = 0; r < 8; ++r) { cv[r] = acc[r]; t0[r] = cv[r] * cv[r]; t1[r] = cv[r] * hbv; }
        red_store(redA, t0, c, rbase);
        red_store(redB, t1, c, rbase);
        __syncthreads();
        red_fold(redA, red2A, tid);
        red_fold(redB, red2B, tid);
        __syncthreads();
        if (tid < 16) {
            float smm = red_final(red2A, tid), smh = red_final(red2B, tid);
            float xn  = fmaxf(sqrtf(rXX[tid]), MINN);
            float mxn = fmaxf(sqrtf(smm), MINN);
            float t = fminf(tanhf(mxn / xn * artanh_(xn)), MAXN);    // mobius_matvec + proj
            float s1 = t / mxn;
            float x2 = t * t;
            float xy = s1 * smh;
            float den = fmaxf(1.f + 2.f * xy + x2 * hbn2, MINN);     // mobius_add
            sA[tid] = (1.f + 2.f * xy + hbn2) / den * s1;
            sB[tid] = (1.f - x2) / den;
        }
        __syncthreads();
#pragma unroll
        for (int r = 0; r < 8; ++r) {
            float o = sA[rbase + r] * cv[r] + sB[rbase + r] * hbv;
            cv[r] = o;
            t0[r] = o * o;
        }
        red_store(redA, t0, c, rbase);
        __syncthreads();
        red_fold(redA, red2A, tid);
        __syncthreads();
        if (tid < 16) {
            float n = fmaxf(sqrtf(red_final(red2A, tid)), MINN);
            float pf = (n > MAXN) ? (MAXN / n) : 1.f;                // final proj
            if (do_logmap) {                                         // logmap0 for ht path
                float nn = fminf(n, MAXN);
                pf *= artanh_(nn) / fmaxf(nn, MINN);
            }
            sF[tid] = pf;
        }
        __syncthreads();
#pragma unroll
        for (int r = 0; r < 8; ++r) {
            int rib = rt * 16 + rbase + r;
            float o = (rib < NROI) ? sF[rbase + r] * cv[r] : 0.f;    // zero pad rows
            size_t idx = ((size_t)blk * PAD + rib) * PAD + cn;
            if (out_bf16) ((__bf16*)outPtr)[idx] = (__bf16)o;
            else          ((float*)outPtr)[idx] = o;
        }
        __syncthreads();
    }
}

// ---- K2: Q_mapped = mobius_matvec(x_q, x_k) per batch, then row softmax -> att (bf16) ----
__global__ __launch_bounds__(256) void k_att(const __bf16* __restrict__ XQ,
                                             const __bf16* __restrict__ XK,
                                             __bf16* __restrict__ ATT) {
    __shared__ __bf16 Wld[PAD * PAD];
    __shared__ __bf16 AldT[16 * PAD];
    __shared__ float redA[128 * 16];
    __shared__ float red2A[256];
    __shared__ float rXX[16], sS1[16], sSum[16];
    int tid = threadIdx.x, wave = tid >> 5, lane = tid & 31, blk = blockIdx.x;
    {   // x_q[b] is a contiguous 32KB bf16 block: raw b128 copy
        const uint4* src = (const uint4*)(XQ + (size_t)blk * PAD * PAD);
        uint4* dst = (uint4*)Wld;
        for (int i = tid; i < PAD * PAD / 8; i += 256) dst[i] = src[i];
    }
    int cn = (lane & 15) + wave * 16;
    int rbase = (lane >> 4) * 8;
    int c = wave * 16 + (lane & 15);
    bool colvalid = cn < NROI;
    for (int rt = 0; rt < 8; ++rt) {
        const __bf16* Atile = XK + ((size_t)blk * PAD + rt * 16) * PAD;
        float xx;
        v8f acc = gemm_tile_bf16(Atile, Wld, AldT, &xx, tid, wave, lane);
        xx += __shfl_xor(xx, 1); xx += __shfl_xor(xx, 2);
        xx += __shfl_xor(xx, 4); xx += __shfl_xor(xx, 8);
        if ((lane & 15) == 0) rXX[tid >> 4] = xx;
        float cv[8], t0[8];
#pragma unroll
        for (int r = 0; r < 8; ++r) { cv[r] = acc[r]; t0[r] = cv[r] * cv[r]; }
        red_store(redA, t0, c, rbase);
        __syncthreads();
        red_fold(redA, red2A, tid);
        __syncthreads();
        if (tid < 16) {
            float xn  = fmaxf(sqrtf(rXX[tid]), MINN);
            float mxn = fmaxf(sqrtf(red_final(red2A, tid)), MINN);
            sS1[tid] = tanhf(mxn / xn * artanh_(xn)) / mxn;
        }
        __syncthreads();
#pragma unroll
        for (int r = 0; r < 8; ++r) {                     // |row| < 1 => exp safe w/o max-sub
            float e = colvalid ? expf(sS1[rbase + r] * cv[r]) : 0.f;
            cv[r] = e;
            t0[r] = e;
        }
        red_store(redA, t0, c, rbase);
        __syncthreads();
        red_fold(redA, red2A, tid);
        __syncthreads();
        if (tid < 16) sSum[tid] = red_final(red2A, tid);
        __syncthreads();
#pragma unroll
        for (int r = 0; r < 8; ++r) {
            int rib = rt * 16 + rbase + r;
            float o = (rib < NROI && colvalid) ? cv[r] / sSum[rbase + r] : 0.f;
            ATT[((size_t)blk * PAD + rib) * PAD + cn] = (__bf16)o;
        }
        __syncthreads();
    }
}

// ---- K3: G = att[b] @ x_v[b] (f32 out); per-column sums of x_v^2 and G^2 ----
__global__ __launch_bounds__(256) void k_A1G(const __bf16* __restrict__ ATT,
                                             const float* __restrict__ XV,
                                             float* __restrict__ G,
                                             float* __restrict__ xnb, float* __restrict__ g2b) {
    __shared__ __bf16 Wld[PAD * PAD];
    __shared__ __bf16 AldT[16 * PAD];
    int tid = threadIdx.x, wave = tid >> 5, lane = tid & 31, blk = blockIdx.x;
    unsigned* WldU = (unsigned*)Wld;
    {   // W[n][k] = x_v[b][k][n] (transposed stage, f32 source)
        int n = tid >> 1, kb = (tid & 1) * 64;
        float cx = 0.f;
        for (int k2 = kb; k2 < kb + 64; k2 += 2) {
            float a = XV[((size_t)blk * PAD + k2)     * PAD + n];
            float b = XV[((size_t)blk * PAD + k2 + 1) * PAD + n];
            cx += a * a + b * b;
            WldU[n * 64 + (k2 >> 1)] = pack_bf16(a, b);
        }
        cx += __shfl_xor(cx, 1);
        if ((tid & 1) == 0) xnb[blk * PAD + n] = cx;      // ||x_v col||^2
    }
    int cn = (lane & 15) + wave * 16;
    int rbase = (lane >> 4) * 8;
    float locG = 0.f;
    for (int rt = 0; rt < 8; ++rt) {
        const __bf16* Atile = ATT + ((size_t)blk * PAD + rt * 16) * PAD;
        v8f acc = gemm_tile_bf16(Atile, Wld, AldT, nullptr, tid, wave, lane);
#pragma unroll
        for (int r = 0; r < 8; ++r) {
            locG += acc[r] * acc[r];
            G[((size_t)blk * PAD + rt * 16 + rbase + r) * PAD + cn] = acc[r];
        }
    }
    locG += __shfl_xor(locG, 16);                         // two lane-halves share a column
    if (lane < 16) g2b[blk * PAD + cn] = locG;            // unique (wave,lane) per column
}

// ---- K4: adj1 = hyp_act(A1 + A1^T), A1[i][j] = s[i]*G[j][i]; bf16 out, all 128 rows ----
__global__ __launch_bounds__(256) void k_adj1(const float* __restrict__ G,
                                              const float* __restrict__ xnb,
                                              const float* __restrict__ g2b,
                                              __bf16* __restrict__ ADJ) {
    __shared__ float Gl[NROI * NROI];
    __shared__ float sS[PAD], rA[PAD], rLA[PAD], rC[PAD];
    int tid = threadIdx.x, blk = blockIdx.x;
    const float* Gb = G + (size_t)blk * PAD * PAD;
    for (int i = tid; i < NROI * 58; i += 256) {          // pair loads of valid 116x116 region
        int r = i / 58, c2 = (i - r * 58) * 2;
        *(float2*)(Gl + r * NROI + c2) = *(const float2*)(Gb + r * PAD + c2);
    }
    if (tid < PAD) {
        float xn  = fmaxf(sqrtf(fmaxf(xnb[blk * PAD + tid], 0.f)), MINN);
        float mxn = fmaxf(sqrtf(fmaxf(g2b[blk * PAD + tid], 0.f)), MINN);
        sS[tid] = tanhf(mxn / xn * artanh_(xn)) / mxn;
        rA[tid] = 0.f;
    }
    __syncthreads();
    for (int f = tid; f < NROI * PAD; f += 256) {         // row sums of S^2
        int i = f >> 7, j = f & 127;
        if (j < NROI) {
            float s = sS[i] * Gl[j * NROI + i] + sS[j] * Gl[i * NROI + j];
            atomicAdd(&rA[i], s * s);
        }
    }
    __syncthreads();
    if (tid < NROI) {
        float n0 = fmaxf(sqrtf(rA[tid]), MINN);
        rLA[tid] = artanh_(n0) / n0;                      // logmap0 factor
        rA[tid] = 0.f;
    }
    __syncthreads();
    for (int f = tid; f < NROI * PAD; f += 256) {         // row sums of relu(la*S)^2
        int i = f >> 7, j = f & 127;
        if (j < NROI) {
            float s = sS[i] * Gl[j * NROI + i] + sS[j] * Gl[i * NROI + j];
            float v = fmaxf(rLA[i] * s, 0.f);
            atomicAdd(&rA[i], v * v);
        }
    }
    __syncthreads();
    if (tid < NROI) {
        float nv = fmaxf(sqrtf(rA[tid]), MINN);
        rC[tid] = fminf(tanhf(nv), MAXN) / nv;            // expmap0 + proj
    }
    __syncthreads();
    unsigned* ADJu = (unsigned*)ADJ;
    for (int f = tid; f < PAD * 64; f += 256) {           // write all rows, zero pads
        int i = f >> 6, j2 = (f & 63) * 2;
        float ox = 0.f, oy = 0.f;
        if (i < NROI) {
            if (j2 < NROI) {
                float s = sS[i] * Gl[j2 * NROI + i] + sS[j2] * Gl[i * NROI + j2];
                ox = rC[i] * fmaxf(rLA[i] * s, 0.f);
            }
            if (j2 + 1 < NROI) {
                float s = sS[i] * Gl[(j2 + 1) * NROI + i] + sS[j2 + 1] * Gl[i * NROI + j2 + 1];
                oy = rC[i] * fmaxf(rLA[i] * s, 0.f);
            }
        }
        ADJu[((size_t)blk * PAD + i) * 64 + (j2 >> 1)] = pack_bf16(ox, oy);
    }
}

// ---- K5: agg = adj1 @ ht ; x_t = logmap0(hyp_act(proj(expmap0(agg)))) -> bf16 ----
__global__ __launch_bounds__(256) void k_agg(const __bf16* __restrict__ ADJ,
                                             const float* __restrict__ HT,
                                             __bf16* __restrict__ XT) {
    __shared__ __bf16 Wld[PAD * PAD];
    __shared__ __bf16 AldT[16 * PAD];
    __shared__ float redA[128 * 16];
    __shared__ float red2A[256];
    __shared__ float sF1[16], sF2[16];
    int tid = threadIdx.x, wave = tid >> 5, lane = tid & 31, blk = blockIdx.x;
    unsigned* WldU = (unsigned*)Wld;
    {   // W[n][k] = ht[b][k][n] (transposed stage, f32 source)
        int n = tid >> 1, kb = (tid & 1) * 64;
        for (int k2 = kb; k2 < kb + 64; k2 += 2) {
            float a = HT[((size_t)blk * PAD + k2)     * PAD + n];
            float b = HT[((size_t)blk * PAD + k2 + 1) * PAD + n];
            WldU[n * 64 + (k2 >> 1)] = pack_bf16(a, b);
        }
    }
    int cn = (lane & 15) + wave * 16;
    int rbase = (lane >> 4) * 8;
    int c = wave * 16 + (lane & 15);
    for (int rt = 0; rt < 8; ++rt) {
        const __bf16* Atile = ADJ + ((size_t)blk * PAD + rt * 16) * PAD;
        v8f acc = gemm_tile_bf16(Atile, Wld, AldT, nullptr, tid, wave, lane);
        float cv[8], t0[8];
#pragma unroll
        for (int r = 0; r < 8; ++r) { cv[r] = acc[r]; t0[r] = cv[r] * cv[r]; }
        red_store(redA, t0, c, rbase);
        __syncthreads();
        red_fold(redA, red2A, tid);
        __syncthreads();
        if (tid < 16) {
            float n1 = fmaxf(sqrtf(red_final(red2A, tid)), MINN);
            float t1 = tanhf(n1);
            float c1 = fminf(t1, MAXN) / n1;              // expmap0 + proj
            float np = fminf(t1, MAXN);
            sF1[tid] = (artanh_(np) / fmaxf(np, MINN)) * c1;  // * logmap0 factor
        }
        __syncthreads();
#pragma unroll
        for (int r = 0; r < 8; ++r) {
            float rl = fmaxf(cv[r], 0.f);                 // relu (sF1 >= 0)
            cv[r] = rl;
            float v = sF1[rbase + r] * rl;
            t0[r] = v * v;
        }
        red_store(redA, t0, c, rbase);
        __syncthreads();
        red_fold(redA, red2A, tid);
        __syncthreads();
        if (tid < 16) {
            float nv = fmaxf(sqrtf(red_final(red2A, tid)), MINN);
            float tt = tanhf(nv);
            float c2 = fminf(tt, MAXN) / nv;              // expmap0 + proj
            float g  = fminf(tt, MAXN);
            float lt = artanh_(g) / fmaxf(g, MINN);       // final logmap0
            sF2[tid] = lt * c2 * sF1[tid];
        }
        __syncthreads();
#pragma unroll
        for (int r = 0; r < 8; ++r) {
            int rib = rt * 16 + rbase + r;
            float o = (rib < NROI) ? sF2[rbase + r] * cv[r] : 0.f;
            XT[((size_t)blk * PAD + rib) * PAD + cn] = (__bf16)o;
        }
        __syncthreads();
    }
}

// ---- K6: community pooling -> conv -> proj(expmap0) -> hyp_linear(Whl,bhl) -> out (B,2) ----
__global__ __launch_bounds__(256) void k_head(const __bf16* __restrict__ XT,
                                              const float* __restrict__ CW, const float* __restrict__ CB,
                                              const float* __restrict__ VW, const float* __restrict__ VB,
                                              const float* __restrict__ WHL,
                                              const float* __restrict__ HBL, const float* __restrict__ HBLN2,
                                              float* __restrict__ OUT) {
    __shared__ float fsum[7 * 64];
    __shared__ float pd[256];
    __shared__ float brl[32];
    const int offs[8] = {0, 20, 36, 54, 68, 84, 96, 116};
    int tid = threadIdx.x, blk = blockIdx.x;
    for (int i = tid; i < 7 * 64; i += 256) fsum[i] = 0.f;
    __syncthreads();
    int of = tid & 63, part = tid >> 6;
    for (int ci = 0; ci < 7; ++ci) {
        float inv = 1.f / (float)(offs[ci + 1] - offs[ci]);
        for (int rr = offs[ci]; rr < offs[ci + 1]; ++rr) {
            const __bf16* xr = XT + ((size_t)blk * PAD + rr) * PAD;
            const float* wr = CW + ((size_t)ci * 64 + of) * NROI;
            int k0 = part * 29, k1 = k0 + 29; if (k1 > NROI) k1 = NROI;
            float s = 0.f;
            for (int k = k0; k < k1; ++k) s += (float)xr[k] * wr[k];
            pd[tid] = s;
            __syncthreads();
            if (tid < 64) {
                float tot = pd[tid] + pd[tid + 64] + pd[tid + 128] + pd[tid + 192] + CB[ci * 64 + tid];
                fsum[ci * 64 + tid] += fmaxf(tot, 0.f) * inv;
            }
            __syncthreads();
        }
    }
    if (tid < 32) {
        float s = VB[tid];
        for (int i = 0; i < 7; ++i)
            for (int f = 0; f < 64; ++f) s += fsum[i * 64 + f] * VW[tid * 448 + i * 64 + f];
        brl[tid] = s;
    }
    __syncthreads();
    if (tid == 0) {
        float nb2 = 0.f;
        for (int o = 0; o < 32; ++o) nb2 += brl[o] * brl[o];
        float nB = fmaxf(sqrtf(nb2), MINN);
        float cB = fminf(tanhf(nB), MAXN) / nB;           // proj(expmap0(brain))
        float p[32];
        for (int o = 0; o < 32; ++o) p[o] = cB * brl[o];
        float xn = fmaxf(fminf(tanhf(nB), MAXN), MINN);   // |p|
        float mx0 = 0.f, mx1 = 0.f;
        for (int o = 0; o < 32; ++o) { mx0 += p[o] * WHL[o]; mx1 += p[o] * WHL[32 + o]; }
        float mxn = fmaxf(sqrtf(mx0 * mx0 + mx1 * mx1), MINN);
        float t = fminf(tanhf(mxn / xn * artanh_(xn)), MAXN);
        float s1 = t / mxn;
        float r0 = s1 * mx0, r1 = s1 * mx1;
        float hl0 = HBL[0], hl1 = HBL[1], y2 = HBLN2[0];
        float xy = r0 * hl0 + r1 * hl1, x2 = t * t;
        float den = fmaxf(1.f + 2.f * xy + x2 * y2, MINN);
        float ca = (1.f + 2.f * xy + y2) / den, cb2 = (1.f - x2) / den;
        float o0 = ca * r0 + cb2 * hl0, o1 = ca * r1 + cb2 * hl1;
        float n = fmaxf(sqrtf(o0 * o0 + o1 * o1), MINN);
        if (n > MAXN) { float fpr = MAXN / n; o0 *= fpr; o1 *= fpr; }
        OUT[blk * 2 + 0] = o0;
        OUT[blk * 2 + 1] = o1;
    }
}

extern "C" void kernel_launch(void* const* d_in, const int* in_sizes, int n_in,
                              void* d_out, int out_size, void* d_ws, size_t ws_size,
                              hipStream_t stream) {
    (void)in_sizes; (void)n_in; (void)out_size; (void)ws_size;
    const float* x   = (const float*)d_in[0];
    // d_in[1]=adj (unused by reference), d_in[2]=ROIs_belong (== arange -> contiguous slices)
    const float* Wq  = (const float*)d_in[3];  const float* bq  = (const float*)d_in[4];
    const float* Wk  = (const float*)d_in[5];  const float* bk  = (const float*)d_in[6];
    const float* Wv  = (const float*)d_in[7];  const float* bv  = (const float*)d_in[8];
    const float* Wg  = (const float*)d_in[9];  const float* bg  = (const float*)d_in[10];
    const float* cW  = (const float*)d_in[11]; const float* cB  = (const float*)d_in[12];
    const float* vW  = (const float*)d_in[13]; const float* vB  = (const float*)d_in[14];
    const float* Whl = (const float*)d_in[15]; const float* bhl = (const float*)d_in[16];
    float* out = (float*)d_out;
    char* ws = (char*)d_ws;

    float* hbs   = (float*)ws;                            // 4*128
    float* hbn2  = hbs + 4 * PAD;                         // 4
    float* hbl   = hbn2 + 4;                              // 2
    float* hbln2 = hbl + 2;                               // 1
    float* xnb   = (float*)(ws + 4096);                   // 512*128
    float* g2b   = xnb + (size_t)NBATCH * PAD;            // 512*128

    const size_t NELT = (size_t)NBATCH * PAD * PAD;       // elements per big buffer
    char* p = ws + (1u << 20);
    __bf16* bxp  = (__bf16*)p;            p += NELT * 2;  // padded x (bf16)
    __bf16* bxq  = (__bf16*)p;            p += NELT * 2;  // x_q; reused for adj1
    __bf16* bxk  = (__bf16*)p;            p += NELT * 2;  // x_k; reused for x_t
    __bf16* batt = (__bf16*)p;            p += NELT * 2;  // att
    float*  bxv  = (float*)p;             p += NELT * 4;  // x_v (f32: transposed consumer)
    float*  bht  = (float*)p;             p += NELT * 4;  // ht  (f32: transposed consumer)
    float*  bG   = (float*)p;                             // G = att @ x_v (f32)

    k_prep<<<1, 32, 0, stream>>>(bq, bk, bv, bg, bhl, hbs, hbn2, hbl, hbln2);
    k_pad<<<(int)(NELT / 512), 256, 0, stream>>>(x, (unsigned*)bxp);
    k_hyp_linear<<<NBATCH, 256, 0, stream>>>(bxp, Wq, hbs + 0 * PAD, hbn2 + 0, bxq, 1, 0);
    k_hyp_linear<<<NBATCH, 256, 0, stream>>>(bxp, Wk, hbs + 1 * PAD, hbn2 + 1, bxk, 1, 0);
    k_hyp_linear<<<NBATCH, 256, 0, stream>>>(bxp, Wv, hbs + 2 * PAD, hbn2 + 2, bxv, 0, 0);
    k_hyp_linear<<<NBATCH, 256, 0, stream>>>(bxp, Wg, hbs + 3 * PAD, hbn2 + 3, bht, 0, 1);
    k_att<<<NBATCH, 256, 0, stream>>>(bxq, bxk, batt);
    k_A1G<<<NBATCH, 256, 0, stream>>>(batt, bxv, bG, xnb, g2b);
    k_adj1<<<NBATCH, 256, 0, stream>>>(bG, xnb, g2b, bxq);     // adj1 -> bxq (x_q dead)
    k_agg<<<NBATCH, 256, 0, stream>>>(bxq, bht, bxk);          // x_t  -> bxk (x_k dead)
    k_head<<<NBATCH, 256, 0, stream>>>(bxk, cW, cB, vW, vB, Whl, hbl, hbln2, out);
}
